// Pointatt_49959059587537
// MI455X (gfx1250) — matrix operand order repaired and checked by reading, not compiled, not run on hardware
//
#include <hip/hip_runtime.h>

#define NN 778
#define FF 128
#define HH 8
#define KP 256                 // 2F
#define NT ((NN + 15) / 16)    // 49 tiles of 16

typedef __attribute__((ext_vector_type(16))) __bf16 v16bf;
typedef __attribute__((ext_vector_type(8)))  float  v8f;

union FragBF { unsigned int u[8]; v16bf v; };

__device__ __forceinline__ unsigned short f2bf(float x) {
  unsigned int u = __float_as_uint(x);
  unsigned int r = u + 0x7FFFu + ((u >> 16) & 1u);   // round-to-nearest-even
  return (unsigned short)(r >> 16);
}

__device__ __forceinline__ v8f wmma_bf16(FragBF a, FragBF b, v8f c) {
  return __builtin_amdgcn_wmma_f32_16x16x32_bf16(false, a.v, false, b.v,
                                                 (short)0, c, false, false);
}

// ---------------------------------------------------------------------------
// Precompute: Lq/Lv/Rk projections (N x F each)
// ---------------------------------------------------------------------------
__global__ __launch_bounds__(256) void proj_qvk(
    const float* __restrict__ Lf, const float* __restrict__ Rf,
    const float* __restrict__ Wq, const float* __restrict__ bq,
    const float* __restrict__ Wv, const float* __restrict__ bv,
    const float* __restrict__ Wk, const float* __restrict__ bk,
    float* __restrict__ Lq, float* __restrict__ Lv, float* __restrict__ Rk)
{
  int idx = blockIdx.x * 256 + threadIdx.x;
  if (idx >= 3 * NN * FF) return;
  int which = idx / (NN * FF);
  int rem   = idx % (NN * FF);
  int n = rem >> 7, c = rem & 127;
  const float* in = (which == 2) ? Rf : Lf;
  const float* W  = (which == 0) ? Wq : (which == 1) ? Wv : Wk;
  const float* b  = (which == 0) ? bq : (which == 1) ? bv : bk;
  float a = b[c];
  for (int k = 0; k < 128; ++k) a += in[n * 128 + k] * W[k * 128 + c];
  float* o = (which == 0) ? Lq : (which == 1) ? Lv : Rk;
  o[rem] = a;
}

// ---------------------------------------------------------------------------
// Precompute: PA = left_pos@W1 + b1 ; PC = right_pos@W1   (N x 256 each)
// ---------------------------------------------------------------------------
__global__ __launch_bounds__(256) void proj_pos(
    const float* __restrict__ lp, const float* __restrict__ rp,
    const float* __restrict__ W1, const float* __restrict__ b1,
    float* __restrict__ PA, float* __restrict__ PC)
{
  int idx = blockIdx.x * 256 + threadIdx.x;
  if (idx >= 2 * NN * KP) return;
  int which = idx / (NN * KP);
  int rem   = idx % (NN * KP);
  int n = rem >> 8, k = rem & 255;
  const float* in = (which == 0) ? lp : rp;
  float a = (which == 0) ? b1[k] : 0.f;
  for (int c = 0; c < 128; ++c) a += in[n * 128 + c] * W1[c * 256 + k];
  float* o = (which == 0) ? PA : PC;
  o[rem] = a;
}

// ---------------------------------------------------------------------------
// Pack weights as bf16 pairs in B-fragment-friendly (k-pair, n) order
// ---------------------------------------------------------------------------
__global__ __launch_bounds__(256) void pack_weights(
    const float* __restrict__ W2,   // (256,128)
    const float* __restrict__ c1w,  // (H,32,16)
    const float* __restrict__ c2w,  // (H,16,32)
    unsigned int* __restrict__ W2pk,   // [128][128]
    unsigned int* __restrict__ B1pk,   // [H][16][32], rows kk>=8 zero (K pad)
    unsigned int* __restrict__ B2pk)   // [H][16][16]
{
  int idx = blockIdx.x * 256 + threadIdx.x;
  if (idx < 128 * 128) {
    int kk = idx >> 7, n = idx & 127;
    unsigned int lo = f2bf(W2[(2 * kk) * 128 + n]);
    unsigned int hu = f2bf(W2[(2 * kk + 1) * 128 + n]);
    W2pk[idx] = lo | (hu << 16);
  } else if (idx < 128 * 128 + HH * 16 * 32) {
    int r = idx - 128 * 128;
    int h = r >> 9, kk = (r >> 5) & 15, o = r & 31;
    unsigned int pv = 0u;
    if (kk < 8) {
      int d = 2 * kk;
      unsigned int lo = f2bf(c1w[(h * 32 + o) * 16 + d]);
      unsigned int hu = f2bf(c1w[(h * 32 + o) * 16 + d + 1]);
      pv = lo | (hu << 16);
    }
    B1pk[r] = pv;
  } else if (idx < 128 * 128 + HH * 16 * 32 + HH * 16 * 16) {
    int r = idx - 128 * 128 - HH * 16 * 32;
    int h = r >> 8, kk = (r >> 4) & 15, d = r & 15;
    int o = 2 * kk;
    unsigned int lo = f2bf(c2w[(h * 16 + d) * 32 + o]);
    unsigned int hu = f2bf(c2w[(h * 16 + d) * 32 + o + 1]);
    B2pk[r] = lo | (hu << 16);
  }
}

__global__ __launch_bounds__(256) void zero_f(float* __restrict__ p, int n) {
  int i = blockIdx.x * 256 + threadIdx.x;
  if (i < n) p[i] = 0.f;
}

// ---------------------------------------------------------------------------
// Pair pass. MODE 0: block = 16 i's x 1 j; accumulate sum_i exp(sim) -> S[j,c]
//           MODE 1: block = 1 i x 16 j's; accumulate sum_j attn*(Lv+rel) -> AGG[i,c]
// 256 threads = 8 waves; wave w owns channels [16w,16w+16) == head w.
// ---------------------------------------------------------------------------
template<int MODE>
__global__ __launch_bounds__(256) void pair_pass(
    const float* __restrict__ PA, const float* __restrict__ PC,
    const unsigned int* __restrict__ W2pk,
    const unsigned int* __restrict__ B1pk,
    const unsigned int* __restrict__ B2pk,
    const float* __restrict__ pos_b2,
    const float* __restrict__ c1b, const float* __restrict__ c2b,
    const float* __restrict__ Lq, const float* __restrict__ Rk,
    const float* __restrict__ Lv,
    const float* __restrict__ Sin,
    float* __restrict__ OUT)
{
  const int tid  = threadIdx.x;
  const int w    = tid >> 5;        // wave == head
  const int lane = tid & 31;
  const bool hi  = lane >= 16;
  const int ln   = lane & 15;
  const int mbase = blockIdx.x * 16;   // MODE0: i-base, MODE1: j-base
  const int fixed = blockIdx.y;        // MODE0: j,      MODE1: i

  __shared__ unsigned short hid16[16][KP];   // relu(PA-PC) bf16
  __shared__ float          qk[16][FF];      // Lq[i]-Rk[j]
  __shared__ unsigned short xs[HH][16][16];  // per-head x bf16
  __shared__ unsigned short hs[HH][16][32];  // per-head hid2 bf16

  // ---- stage hidden (16 x 256) and qk (16 x 128), invalid rows -> 0 ----
  for (int s = 0; s < 16; ++s) {
    int e = tid + 256 * s;
    int m = e >> 8, k = e & 255;
    int mi = mbase + m;
    float val = 0.f;
    if (mi < NN) {
      int ii = (MODE == 0) ? mi : fixed;
      int jj = (MODE == 0) ? fixed : mi;
      val = PA[ii * KP + k] - PC[jj * KP + k];
      val = val > 0.f ? val : 0.f;
    }
    hid16[m][k] = f2bf(val);
  }
  for (int s = 0; s < 8; ++s) {
    int e = tid + 256 * s;
    int m = e >> 7, c = e & 127;
    int mi = mbase + m;
    float val = 0.f;
    if (mi < NN) {
      int ii = (MODE == 0) ? mi : fixed;
      int jj = (MODE == 0) ? fixed : mi;
      val = Lq[ii * FF + c] - Rk[jj * FF + c];
    }
    qk[m][c] = val;
  }
  __syncthreads();

  const unsigned int* hidU = (const unsigned int*)&hid16[0][0];  // [16][128]
  const int nch = w * 16 + ln;

  // ---- rel = hidden @ W2 + b2 : M=16 pairs, N=16 ch (this wave), K=256 ----
  v8f rel;
  #pragma unroll
  for (int r = 0; r < 8; ++r) rel[r] = 0.f;
  #pragma unroll
  for (int ks = 0; ks < 8; ++ks) {
    FragBF a, b;
    #pragma unroll
    for (int q = 0; q < 8; ++q) {
      int kkA = q + (hi ? 8 : 0);                          // A 16x32 bf16 layout
      a.u[q] = hidU[ln * 128 + ks * 16 + kkA];
      int kkB = q + (hi ? 8 : 0);                          // B 32x16 bf16 layout
      b.u[q] = W2pk[(ks * 16 + kkB) * FF + nch];
    }
    rel = wmma_bf16(a, b, rel);
  }
  {
    float b2v = pos_b2[nch];
    #pragma unroll
    for (int r = 0; r < 8; ++r) rel[r] += b2v;
  }

  // ---- x = rel + (Lq-Rk), to LDS as bf16 (C-layout -> A-layout bounce) ----
  #pragma unroll
  for (int r = 0; r < 8; ++r) {
    int m = r + (hi ? 8 : 0);
    xs[w][m][ln] = f2bf(rel[r] + qk[m][nch]);
  }
  __syncthreads();

  // ---- conv1: per-head D=16 (K padded to 32) -> 32 channels, 2 tiles ----
  const unsigned int* xu = (const unsigned int*)&xs[w][0][0];  // [16][8]
  v8f h0, h1;
  {
    float bb0 = c1b[w * 32 + ln];
    float bb1 = c1b[w * 32 + 16 + ln];
    #pragma unroll
    for (int r = 0; r < 8; ++r) { h0[r] = bb0; h1[r] = bb1; }
  }
  {
    FragBF a, b0, b1;
    #pragma unroll
    for (int q = 0; q < 8; ++q) {
      int kkA = q + (hi ? 8 : 0);
      a.u[q] = (kkA < 8) ? xu[ln * 8 + kkA] : 0u;          // K>=16 is zero pad
      int kkB = q + (hi ? 8 : 0);
      b0.u[q] = B1pk[(w * 16 + kkB) * 32 + ln];
      b1.u[q] = B1pk[(w * 16 + kkB) * 32 + 16 + ln];
    }
    h0 = wmma_bf16(a, b0, h0);
    h1 = wmma_bf16(a, b1, h1);
  }
  #pragma unroll
  for (int r = 0; r < 8; ++r) {
    int m = r + (hi ? 8 : 0);
    float v0 = h0[r] > 0.f ? h0[r] : 0.f;
    float v1 = h1[r] > 0.f ? h1[r] : 0.f;
    hs[w][m][ln]      = f2bf(v0);
    hs[w][m][16 + ln] = f2bf(v1);
  }
  __syncthreads();

  // ---- conv2: K=32 exact -> sim (16 channels) ----
  const unsigned int* hu = (const unsigned int*)&hs[w][0][0];  // [16][16]
  v8f sim;
  {
    float bb = c2b[w * 16 + ln];
    #pragma unroll
    for (int r = 0; r < 8; ++r) sim[r] = bb;
  }
  {
    FragBF a, b;
    #pragma unroll
    for (int q = 0; q < 8; ++q) {
      int kkA = q + (hi ? 8 : 0);
      a.u[q] = hu[ln * 16 + kkA];
      int kkB = q + (hi ? 8 : 0);
      b.u[q] = B2pk[(w * 16 + kkB) * 16 + ln];
    }
    sim = wmma_bf16(a, b, sim);
  }

  // ---- reduce over the 16 pairs in this block ----
  float acc = 0.f;
  if (MODE == 0) {
    #pragma unroll
    for (int r = 0; r < 8; ++r) {
      int m = r + (hi ? 8 : 0);
      if (mbase + m < NN) acc += __expf(sim[r]);
    }
  } else {
    #pragma unroll
    for (int r = 0; r < 8; ++r) {
      int m = r + (hi ? 8 : 0);
      int jj = mbase + m;
      if (jj < NN) {
        float attn = __expf(sim[r]) / Sin[jj * FF + nch];
        acc += attn * (Lv[jj * FF + nch] + rel[r]);
      }
    }
  }
  acc += __shfl_xor(acc, 16, 32);            // combine wave32 halves (m and m+8)
  if (lane < 16) atomicAdd(&OUT[fixed * FF + w * 16 + lane], acc);
}

// ---------------------------------------------------------------------------
// Final: y = Lf + AGG ; out = y + fc2(relu(fc1(LN(y))))
// One block per row, 256 threads.
// ---------------------------------------------------------------------------
__global__ __launch_bounds__(256) void final_ffn(
    const float* __restrict__ Lf, const float* __restrict__ AGG,
    const float* __restrict__ ln_g, const float* __restrict__ ln_b,
    const float* __restrict__ W1, const float* __restrict__ b1,
    const float* __restrict__ W2, const float* __restrict__ b2,
    float* __restrict__ out)
{
  int n = blockIdx.x, t = threadIdx.x;
  __shared__ float y[128], yn[128], hid[256];
  __shared__ float mu_s, rstd_s;
  if (t < 128) y[t] = Lf[n * 128 + t] + AGG[n * 128 + t];
  __syncthreads();
  if (t == 0) {
    float s = 0.f, s2 = 0.f;
    for (int c = 0; c < 128; ++c) { s += y[c]; s2 += y[c] * y[c]; }
    float mu = s * (1.f / 128.f);
    float var = s2 * (1.f / 128.f) - mu * mu;
    mu_s = mu;
    rstd_s = rsqrtf(var + 1e-6f);
  }
  __syncthreads();
  if (t < 128) yn[t] = (y[t] - mu_s) * rstd_s * ln_g[t] + ln_b[t];
  __syncthreads();
  {
    float a = b1[t];
    for (int c = 0; c < 128; ++c) a += yn[c] * W1[c * 256 + t];
    hid[t] = a > 0.f ? a : 0.f;
  }
  __syncthreads();
  if (t < 128) {
    float a = b2[t];
    for (int o = 0; o < 256; ++o) a += hid[o] * W2[o * 128 + t];
    out[n * 128 + t] = y[t] + a;
  }
}

// ---------------------------------------------------------------------------
extern "C" void kernel_launch(void* const* d_in, const int* in_sizes, int n_in,
                              void* d_out, int out_size, void* d_ws, size_t ws_size,
                              hipStream_t stream) {
  (void)in_sizes; (void)n_in; (void)out_size; (void)ws_size;
  const float* Lf      = (const float*)d_in[0];
  const float* Rf      = (const float*)d_in[1];
  const float* lpos    = (const float*)d_in[2];
  const float* rpos    = (const float*)d_in[3];
  const float* pos_W1  = (const float*)d_in[4];
  const float* pos_b1  = (const float*)d_in[5];
  const float* pos_W2  = (const float*)d_in[6];
  const float* pos_b2  = (const float*)d_in[7];
  const float* Wq      = (const float*)d_in[8];
  const float* bq      = (const float*)d_in[9];
  const float* Wv      = (const float*)d_in[10];
  const float* bv      = (const float*)d_in[11];
  const float* Wk      = (const float*)d_in[12];
  const float* bk      = (const float*)d_in[13];
  const float* conv1_w = (const float*)d_in[14];
  const float* conv1_b = (const float*)d_in[15];
  const float* conv2_w = (const float*)d_in[16];
  const float* conv2_b = (const float*)d_in[17];
  const float* ln_g    = (const float*)d_in[18];
  const float* ln_b    = (const float*)d_in[19];
  const float* ff_W1   = (const float*)d_in[20];
  const float* ff_b1   = (const float*)d_in[21];
  const float* ff_W2   = (const float*)d_in[22];
  const float* ff_b2   = (const float*)d_in[23];

  const int NF = NN * FF;           // 99584
  const int NK = NN * KP;           // 199168
  float* ws  = (float*)d_ws;
  float* Lq  = ws;
  float* Lv  = Lq + NF;
  float* Rk  = Lv + NF;
  float* PA  = Rk + NF;
  float* PC  = PA + NK;
  float* S   = PC + NK;
  float* AGG = S + NF;
  unsigned int* W2pk = (unsigned int*)(AGG + NF);   // 128*128
  unsigned int* B1pk = W2pk + 128 * 128;            // 8*16*32
  unsigned int* B2pk = B1pk + HH * 16 * 32;         // 8*16*16

  pack_weights<<<88, 256, 0, stream>>>(pos_W2, conv1_w, conv2_w, W2pk, B1pk, B2pk);
  proj_qvk<<<(3 * NF + 255) / 256, 256, 0, stream>>>(Lf, Rf, Wq, bq, Wv, bv, Wk, bk,
                                                     Lq, Lv, Rk);
  proj_pos<<<(2 * NK + 255) / 256, 256, 0, stream>>>(lpos, rpos, pos_W1, pos_b1, PA, PC);
  zero_f<<<(2 * NF + 255) / 256, 256, 0, stream>>>(S, 2 * NF);  // zeros S and AGG

  dim3 grid(NT, NN);
  pair_pass<0><<<grid, 256, 0, stream>>>(PA, PC, W2pk, B1pk, B2pk, pos_b2,
                                         conv1_b, conv2_b, Lq, Rk, Lv, S, S);
  pair_pass<1><<<grid, 256, 0, stream>>>(PA, PC, W2pk, B1pk, B2pk, pos_b2,
                                         conv1_b, conv2_b, Lq, Rk, Lv, S, AGG);

  final_ffn<<<NN, 256, 0, stream>>>(Lf, AGG, ln_g, ln_b, ff_W1, ff_b1, ff_W2, ff_b2,
                                    (float*)d_out);
}
